// WinBinVecFC_18279380812371
// MI455X (gfx1250) — matrix operand order repaired and checked by reading, not compiled, not run on hardware
//
#include <hip/hip_runtime.h>
#include <hip/hip_bf16.h>
#include <stdint.h>

typedef __bf16 bf16;
typedef __attribute__((ext_vector_type(16))) __bf16 v16bf;
typedef __attribute__((ext_vector_type(8)))  float  v8f;

#define BATCH 8192
#define NGRP  430
#define K0    1720
#define K0PAD 1728

static __host__ int cdiv(int a, int b) { return (a + b - 1) / b; }

// ---------------------------------------------------------------------------
// Weight convert + transpose: W [K][N] f32 -> Wt [N][Kpad] bf16 (zero-padded K)
// ---------------------------------------------------------------------------
__global__ void cvt_w_kernel(const float* __restrict__ W, bf16* __restrict__ Wt,
                             int K, int N, int Kpad) {
    int idx = blockIdx.x * blockDim.x + threadIdx.x;
    int total = N * Kpad;
    if (idx >= total) return;
    int n = idx / Kpad;
    int k = idx - n * Kpad;
    float v = (k < K) ? W[(size_t)k * N + n] : 0.0f;
    Wt[idx] = (bf16)v;
}

// ---------------------------------------------------------------------------
// Zero the K-pad columns of H0b (rows x [K0, K0PAD))
// ---------------------------------------------------------------------------
__global__ void zero_pad_kernel(bf16* __restrict__ H0b) {
    int idx = blockIdx.x * blockDim.x + threadIdx.x;
    const int padw = K0PAD - K0;
    int total = BATCH * padw;
    if (idx >= total) return;
    int r = idx / padw;
    int c = K0 + (idx - r * padw);
    H0b[(size_t)r * K0PAD + c] = (bf16)0.0f;
}

// ---------------------------------------------------------------------------
// Batched tiny GEMM: [B,G,25] x [G,25,4] + bg -> H0b [B, G*4] (bf16, stride K0PAD)
// Memory-bound on X (352 MB); one thread per (b,g) pair.
// ---------------------------------------------------------------------------
__global__ void group_gemm_kernel(const float* __restrict__ X,
                                  const float* __restrict__ Wg,
                                  const float* __restrict__ bg,
                                  bf16* __restrict__ H0b) {
    int idx = blockIdx.x * blockDim.x + threadIdx.x;
    if (idx >= BATCH * NGRP) return;
    int b = idx / NGRP;
    int g = idx - b * NGRP;
    const float* x = X + (size_t)idx * 25;
    const float* w = Wg + (size_t)g * 25 * 4;
    float acc0 = bg[g * 4 + 0];
    float acc1 = bg[g * 4 + 1];
    float acc2 = bg[g * 4 + 2];
    float acc3 = bg[g * 4 + 3];
#pragma unroll
    for (int i = 0; i < 25; ++i) {
        float xi = x[i];
        acc0 += xi * w[i * 4 + 0];
        acc1 += xi * w[i * 4 + 1];
        acc2 += xi * w[i * 4 + 2];
        acc3 += xi * w[i * 4 + 3];
    }
    bf16* out = H0b + (size_t)b * K0PAD + g * 4;
    out[0] = (bf16)acc0;
    out[1] = (bf16)acc1;
    out[2] = (bf16)acc2;
    out[3] = (bf16)acc3;
}

// ---------------------------------------------------------------------------
// WMMA fragment loaders (wave32; contiguous k = 16*half + s in both A and B, not the ISA table order)
// ---------------------------------------------------------------------------
__device__ __forceinline__ v16bf load_a_frag(const bf16* __restrict__ A, int lda,
                                             int m0, int kb, int lane) {
    int half = lane >> 4;
    int row  = m0 + (lane & 15);
    const bf16* p = A + (size_t)row * lda + kb;
    v16bf f;
#pragma unroll
    for (int v = 0; v < 8; ++v) {
        int k = 2 * v + 16 * half;  // K of even element
        f[2 * v]     = p[k];
        f[2 * v + 1] = p[k + 1];
    }
    return f;
}

__device__ __forceinline__ v16bf load_b_frag(const bf16* __restrict__ Wt, int ldb,
                                             int n0, int kb, int lane) {
    int half = lane >> 4;
    int col  = n0 + (lane & 15);
    const bf16* p = Wt + (size_t)col * ldb + kb + 16 * half;
    v16bf f;
#pragma unroll
    for (int i = 0; i < 16; ++i) f[i] = p[i];
    return f;
}

// ---------------------------------------------------------------------------
// Big WMMA GEMM for FC3/4/5: wave tile 32x64 (2 A-frags x 4 B-frags = 8
// independent WMMA accumulate chains), block = 8 waves (4x2) -> 128x128 tile.
// Raises arithmetic intensity ~2.7x over 16x16/wave and gives the XDL pipe
// 8-way ILP to cover WMMA latency. OUT_MODE 0: f32 (pre-BN). 1: bf16.
// ---------------------------------------------------------------------------
template <int OUT_MODE>
__global__ void gemm_wmma_big_kernel(const bf16* __restrict__ A,
                                     const bf16* __restrict__ Wt,
                                     const float* __restrict__ bias,
                                     float* __restrict__ Cf,
                                     bf16* __restrict__ Cb,
                                     int N, int Kpad) {
    int lane = threadIdx.x & 31;
    int w    = threadIdx.x >> 5;
    int m0   = blockIdx.y * 128 + (w >> 1) * 32;  // wave covers rows [m0, m0+32)
    int n0   = blockIdx.x * 128 + (w & 1) * 64;   // wave covers cols [n0, n0+64)

    v8f acc[2][4];
#pragma unroll
    for (int i = 0; i < 2; ++i)
#pragma unroll
        for (int j = 0; j < 4; ++j) acc[i][j] = v8f{};

    for (int kb = 0; kb < Kpad; kb += 32) {
        v16bf a0 = load_a_frag(A, Kpad, m0,      kb, lane);
        v16bf a1 = load_a_frag(A, Kpad, m0 + 16, kb, lane);
        v16bf b0 = load_b_frag(Wt, Kpad, n0,      kb, lane);
        v16bf b1 = load_b_frag(Wt, Kpad, n0 + 16, kb, lane);
        v16bf b2 = load_b_frag(Wt, Kpad, n0 + 32, kb, lane);
        v16bf b3 = load_b_frag(Wt, Kpad, n0 + 48, kb, lane);
        acc[0][0] = __builtin_amdgcn_wmma_f32_16x16x32_bf16(false, a0, false, b0, (short)0, acc[0][0], false, false);
        acc[0][1] = __builtin_amdgcn_wmma_f32_16x16x32_bf16(false, a0, false, b1, (short)0, acc[0][1], false, false);
        acc[0][2] = __builtin_amdgcn_wmma_f32_16x16x32_bf16(false, a0, false, b2, (short)0, acc[0][2], false, false);
        acc[0][3] = __builtin_amdgcn_wmma_f32_16x16x32_bf16(false, a0, false, b3, (short)0, acc[0][3], false, false);
        acc[1][0] = __builtin_amdgcn_wmma_f32_16x16x32_bf16(false, a1, false, b0, (short)0, acc[1][0], false, false);
        acc[1][1] = __builtin_amdgcn_wmma_f32_16x16x32_bf16(false, a1, false, b1, (short)0, acc[1][1], false, false);
        acc[1][2] = __builtin_amdgcn_wmma_f32_16x16x32_bf16(false, a1, false, b2, (short)0, acc[1][2], false, false);
        acc[1][3] = __builtin_amdgcn_wmma_f32_16x16x32_bf16(false, a1, false, b3, (short)0, acc[1][3], false, false);
    }

    int half = lane >> 4;
#pragma unroll
    for (int j = 0; j < 4; ++j) {
        int col  = n0 + j * 16 + (lane & 15);
        float bv = bias[col];
#pragma unroll
        for (int i = 0; i < 2; ++i) {
#pragma unroll
            for (int r = 0; r < 8; ++r) {
                int row = m0 + i * 16 + r + 8 * half;
                float v = acc[i][j][r] + bv;
                if (OUT_MODE == 0) Cf[(size_t)row * N + col] = v;
                else               Cb[(size_t)row * N + col] = (bf16)v;
            }
        }
    }
}

// ---------------------------------------------------------------------------
// Small WMMA GEMM (FC6/FC7): 16x16 per wave, block tile 64x32.
// ---------------------------------------------------------------------------
template <int OUT_MODE>
__global__ void gemm_wmma_small_kernel(const bf16* __restrict__ A,
                                       const bf16* __restrict__ Wt,
                                       const float* __restrict__ bias,
                                       float* __restrict__ Cf,
                                       bf16* __restrict__ Cb,
                                       int N, int Kpad) {
    int lane = threadIdx.x & 31;
    int w    = threadIdx.x >> 5;
    int m0   = blockIdx.y * 64 + (w >> 1) * 16;
    int n0   = blockIdx.x * 32 + (w & 1) * 16;

    v8f acc = {};
    for (int kb = 0; kb < Kpad; kb += 32) {
        v16bf a = load_a_frag(A, Kpad, m0, kb, lane);
        v16bf b = load_b_frag(Wt, Kpad, n0, kb, lane);
        acc = __builtin_amdgcn_wmma_f32_16x16x32_bf16(
            false, a, false, b, (short)0, acc, false, false);
    }

    int half = lane >> 4;
    int col  = n0 + (lane & 15);
    float bv = bias[col];
#pragma unroll
    for (int r = 0; r < 8; ++r) {
        int row = m0 + r + 8 * half;
        float v = acc[r] + bv;
        if (OUT_MODE == 0) Cf[(size_t)row * N + col] = v;
        else               Cb[(size_t)row * N + col] = (bf16)v;
    }
}

// ---------------------------------------------------------------------------
// BatchNorm (training-mode batch stats) over M=8192 rows, per column.
// Block covers 64 columns; 4 row-partitions per column; 2-pass over f32 input,
// writes normalized bf16 for the next GEMM.
// ---------------------------------------------------------------------------
__global__ void bn_kernel(const float* __restrict__ Hf,
                          const float* __restrict__ gamma,
                          const float* __restrict__ beta,
                          bf16* __restrict__ Hb, int N) {
    __shared__ float s_sum[256];
    __shared__ float s_sq[256];
    __shared__ float s_scale[64];
    __shared__ float s_shift[64];

    int tid  = threadIdx.x;
    int col  = blockIdx.x * 64 + (tid & 63);
    int part = tid >> 6;

    float sum = 0.0f, sq = 0.0f;
    for (int r = part; r < BATCH; r += 4) {
        float x = Hf[(size_t)r * N + col];
        sum += x;
        sq  += x * x;
    }
    s_sum[tid] = sum;
    s_sq[tid]  = sq;
    __syncthreads();

    if (part == 0) {
        float s = s_sum[tid] + s_sum[tid + 64] + s_sum[tid + 128] + s_sum[tid + 192];
        float q = s_sq[tid]  + s_sq[tid + 64]  + s_sq[tid + 128]  + s_sq[tid + 192];
        float mean = s * (1.0f / BATCH);
        float var  = q * (1.0f / BATCH) - mean * mean;
        float rstd = rsqrtf(var + 1e-5f);
        float sc   = gamma[col] * rstd;
        s_scale[tid] = sc;
        s_shift[tid] = beta[col] - mean * sc;
    }
    __syncthreads();

    float sc = s_scale[tid & 63];
    float sh = s_shift[tid & 63];
    for (int r = part; r < BATCH; r += 4) {
        Hb[(size_t)r * N + col] = (bf16)(Hf[(size_t)r * N + col] * sc + sh);
    }
}

// ---------------------------------------------------------------------------
// Final FC8: [8192,32](bf16) x [32,2](f32) + b8 -> out [8192,2] f32
// ---------------------------------------------------------------------------
__global__ void fc8_kernel(const bf16* __restrict__ H,
                           const float* __restrict__ W8,
                           const float* __restrict__ b8,
                           float* __restrict__ out) {
    int row = blockIdx.x * blockDim.x + threadIdx.x;
    if (row >= BATCH) return;
    const bf16* h = H + (size_t)row * 32;
    float a0 = b8[0], a1 = b8[1];
#pragma unroll
    for (int k = 0; k < 32; ++k) {
        float x = (float)h[k];
        a0 += x * W8[k * 2 + 0];
        a1 += x * W8[k * 2 + 1];
    }
    out[row * 2 + 0] = a0;
    out[row * 2 + 1] = a1;
}

// ---------------------------------------------------------------------------
// Launcher
// ---------------------------------------------------------------------------
extern "C" void kernel_launch(void* const* d_in, const int* in_sizes, int n_in,
                              void* d_out, int out_size, void* d_ws, size_t ws_size,
                              hipStream_t stream) {
    (void)in_sizes; (void)n_in; (void)out_size; (void)ws_size;

    const float* X   = (const float*)d_in[0];
    const float* Wg  = (const float*)d_in[1];
    const float* bg  = (const float*)d_in[2];
    const float* W3  = (const float*)d_in[3];
    const float* b3  = (const float*)d_in[4];
    const float* g3  = (const float*)d_in[5];
    const float* be3 = (const float*)d_in[6];
    const float* W4  = (const float*)d_in[7];
    const float* b4  = (const float*)d_in[8];
    const float* g4  = (const float*)d_in[9];
    const float* be4 = (const float*)d_in[10];
    const float* W5  = (const float*)d_in[11];
    const float* b5  = (const float*)d_in[12];
    const float* g5  = (const float*)d_in[13];
    const float* be5 = (const float*)d_in[14];
    const float* W6  = (const float*)d_in[15];
    const float* b6  = (const float*)d_in[16];
    const float* W7  = (const float*)d_in[17];
    const float* b7  = (const float*)d_in[18];
    const float* W8  = (const float*)d_in[19];
    const float* b8  = (const float*)d_in[20];
    float* out = (float*)d_out;

    // Workspace carve (256B aligned)
    uint8_t* ws = (uint8_t*)d_ws;
    size_t off = 0;
    auto carve = [&](size_t bytes) -> void* {
        void* p = ws + off;
        off = (off + bytes + 255) & ~(size_t)255;
        return p;
    };

    bf16*  H0b = (bf16*)carve((size_t)BATCH * K0PAD * 2);
    bf16*  Wt3 = (bf16*)carve((size_t)1024 * K0PAD * 2);
    float* H1f = (float*)carve((size_t)BATCH * 1024 * 4);
    bf16*  H1b = (bf16*)carve((size_t)BATCH * 1024 * 2);
    bf16*  Wt4 = (bf16*)carve((size_t)512 * 1024 * 2);
    float* H2f = (float*)carve((size_t)BATCH * 512 * 4);
    bf16*  H2b = (bf16*)carve((size_t)BATCH * 512 * 2);
    bf16*  Wt5 = (bf16*)carve((size_t)256 * 512 * 2);
    float* H3f = (float*)carve((size_t)BATCH * 256 * 4);
    bf16*  H3b = (bf16*)carve((size_t)BATCH * 256 * 2);
    bf16*  Wt6 = (bf16*)carve((size_t)64 * 256 * 2);
    bf16*  H4b = (bf16*)carve((size_t)BATCH * 64 * 2);
    bf16*  Wt7 = (bf16*)carve((size_t)32 * 64 * 2);
    bf16*  H5b = (bf16*)carve((size_t)BATCH * 32 * 2);

    const int TB = 256;

    // Weight conversions (f32 [K][N] -> bf16 [N][Kpad])
    cvt_w_kernel<<<cdiv(1024 * K0PAD, TB), TB, 0, stream>>>(W3, Wt3, K0, 1024, K0PAD);
    cvt_w_kernel<<<cdiv(512 * 1024, TB),  TB, 0, stream>>>(W4, Wt4, 1024, 512, 1024);
    cvt_w_kernel<<<cdiv(256 * 512, TB),   TB, 0, stream>>>(W5, Wt5, 512, 256, 512);
    cvt_w_kernel<<<cdiv(64 * 256, TB),    TB, 0, stream>>>(W6, Wt6, 256, 64, 256);
    cvt_w_kernel<<<cdiv(32 * 64, TB),     TB, 0, stream>>>(W7, Wt7, 64, 32, 64);

    // H0: batched micro-GEMM (X stream is the bandwidth-dominant step)
    zero_pad_kernel<<<cdiv(BATCH * (K0PAD - K0), TB), TB, 0, stream>>>(H0b);
    group_gemm_kernel<<<cdiv(BATCH * NGRP, TB), TB, 0, stream>>>(X, Wg, bg, H0b);

    // FC3 + BN3  (M=8192, N=1024, K=1728)
    {
        dim3 grid(1024 / 128, BATCH / 128);
        gemm_wmma_big_kernel<0><<<grid, TB, 0, stream>>>(H0b, Wt3, b3, H1f, nullptr, 1024, K0PAD);
        bn_kernel<<<1024 / 64, TB, 0, stream>>>(H1f, g3, be3, H1b, 1024);
    }
    // FC4 + BN4  (N=512, K=1024)
    {
        dim3 grid(512 / 128, BATCH / 128);
        gemm_wmma_big_kernel<0><<<grid, TB, 0, stream>>>(H1b, Wt4, b4, H2f, nullptr, 512, 1024);
        bn_kernel<<<512 / 64, TB, 0, stream>>>(H2f, g4, be4, H2b, 512);
    }
    // FC5 + BN5  (N=256, K=512)
    {
        dim3 grid(256 / 128, BATCH / 128);
        gemm_wmma_big_kernel<0><<<grid, TB, 0, stream>>>(H2b, Wt5, b5, H3f, nullptr, 256, 512);
        bn_kernel<<<256 / 64, TB, 0, stream>>>(H3f, g5, be5, H3b, 256);
    }
    // FC6 (no BN) -> bf16  (N=64, K=256)
    {
        dim3 grid(64 / 32, BATCH / 64);
        gemm_wmma_small_kernel<1><<<grid, TB, 0, stream>>>(H3b, Wt6, b6, nullptr, H4b, 64, 256);
    }
    // FC7 (no BN) -> bf16  (N=32, K=64)
    {
        dim3 grid(32 / 32, BATCH / 64);
        gemm_wmma_small_kernel<1><<<grid, TB, 0, stream>>>(H4b, Wt7, b7, nullptr, H5b, 32, 64);
    }
    // FC8 -> f32 output
    fc8_kernel<<<cdiv(BATCH, TB), TB, 0, stream>>>(H5b, W8, b8, out);
}
